// MultiHeadAttention_43636867727947
// MI455X (gfx1250) — matrix operand order repaired and checked by reading, not compiled, not run on hardware
//
#include <hip/hip_runtime.h>
#include <hip/hip_bf16.h>

// ---------------------------------------------------------------------------
// MHA forward for MI455X (gfx1250): bf16 WMMA + TDM (tensor_load_to_lds)
//   B=2, S=2048, H=2048, NH=16, HD=128
// ---------------------------------------------------------------------------
#define B_  2
#define S_  2048
#define H_  2048
#define NH_ 16
#define HD_ 128

typedef __bf16 bf16_t;
typedef __attribute__((ext_vector_type(16))) __bf16 v16bf;
typedef __attribute__((ext_vector_type(8)))  __bf16 v8bf;
typedef __attribute__((ext_vector_type(4)))  __bf16 v4bf;
typedef __attribute__((ext_vector_type(8)))  float  v8f;
typedef __attribute__((ext_vector_type(4)))  unsigned v4u;
typedef __attribute__((ext_vector_type(8)))  int      v8i_;
typedef __attribute__((ext_vector_type(4)))  int      v4i_;

__device__ __forceinline__ v8f wmma_bf16(v16bf a, v16bf b, v8f c) {
  return __builtin_amdgcn_wmma_f32_16x16x32_bf16(
      false, a, false, b, (short)0, c, false, false);
}

// A-fragment (16x32 row-major tile): lanes 0-15 row=lane, K0..7 / K16..23
__device__ __forceinline__ v16bf load_frag_a(const bf16_t* p, int ld, int lane) {
  const int r  = lane & 15;
  const int kh = (lane >> 4) << 3;
  const bf16_t* q = p + r * ld + kh;
  union { v16bf v; v8bf h[2]; } u;
  u.h[0] = *(const v8bf*)(q);
  u.h[1] = *(const v8bf*)(q + 16);
  return u.v;
}

// B-fragment (32x16) from a TRANSPOSED tile stored [N][K]:
// lanes 0-15: col=lane, K0..7 / K16..23 ; lanes 16-31: col=lane-16, K8..15 / K24..31
__device__ __forceinline__ v16bf load_frag_b(const bf16_t* p, int ld, int lane) {
  const int c  = lane & 15;
  const int kh = (lane >> 4) << 3;
  const bf16_t* q = p + c * ld + kh;
  union { v16bf v; v8bf h[2]; } u;
  u.h[0] = *(const v8bf*)(q);
  u.h[1] = *(const v8bf*)(q + 16);
  return u.v;
}

// ---------------------------------------------------------------------------
// TDM: DMA a 2-D bf16 tile (tw x th elements, row stride `stride` elements)
// from global memory into LDS (contiguous [th][tw]).  D# per ISA §8.3/§8.4.
// Issue from ONE wave per workgroup; EXEC is ignored by TDM.
// ---------------------------------------------------------------------------
__device__ __forceinline__ void tdm_load_2d(void* ldsp, const bf16_t* g,
                                            unsigned tw, unsigned th_,
                                            unsigned stride) {
  const unsigned lds_off = (unsigned)(size_t)ldsp;           // LDS byte offset
  const unsigned long long ga = (unsigned long long)(size_t)g;
  // group0: count=1 | lds_addr | global_addr[56:0] | type=2 (bits 127:126)
  v4u g0 = { 1u, lds_off, (unsigned)ga,
             (unsigned)((ga >> 32) & 0x01FFFFFFull) | (2u << 30) };
  // group1: data_size=2B; tensor_dim0=tw (tile spans full slice -> no OOB);
  // tensor_dim1=th; tile_dim0=tw; tile_dim1=th; tensor_dim0_stride=stride.
  v8i_ g1;
  g1[0] = (int)(1u << 16);                        // data_size=1 -> 2 bytes
  g1[1] = (int)((tw & 0xFFFFu) << 16);            // tensor_dim0[15:0]
  g1[2] = (int)((tw >> 16) | ((th_ & 0xFFFFu) << 16)); // dim0 hi | dim1 lo
  g1[3] = (int)((th_ >> 16) | (tw << 16));        // dim1 hi | tile_dim0
  g1[4] = (int)th_;                               // tile_dim1 (tile_dim2=0)
  g1[5] = (int)stride;                            // tensor_dim0_stride lo32
  g1[6] = 0;
  g1[7] = 0;
  const v4i_ z4 = {0, 0, 0, 0};
#if defined(__clang_major__) && __clang_major__ >= 23
  const v8i_ z8 = {0, 0, 0, 0, 0, 0, 0, 0};
  __builtin_amdgcn_tensor_load_to_lds(g0, g1, z4, z4, z8, 0);
#else
  __builtin_amdgcn_tensor_load_to_lds(g0, g1, z4, z4, 0);
#endif
}

// ---------------------------------------------------------------------------
// fp32 -> bf16 bulk convert (vectorized)
// ---------------------------------------------------------------------------
__global__ __launch_bounds__(256) void convert_bf16x4(
    const float4* __restrict__ src, bf16_t* __restrict__ dst, int n4) {
  int i = blockIdx.x * blockDim.x + threadIdx.x;
  if (i < n4) {
    float4 f = src[i];
    v4bf o;
    o[0] = (bf16_t)f.x; o[1] = (bf16_t)f.y;
    o[2] = (bf16_t)f.z; o[3] = (bf16_t)f.w;
    *(v4bf*)(dst + (size_t)i * 4) = o;
  }
}

// ---------------------------------------------------------------------------
// fp32 W[K][N] -> bf16 Wt[N][K]  (LDS tile transpose)
// ---------------------------------------------------------------------------
__global__ __launch_bounds__(256) void convert_transpose(
    const float* __restrict__ W, bf16_t* __restrict__ Wt, int K, int N) {
  __shared__ bf16_t tile[32][40];
  const int n0 = blockIdx.x * 32;
  const int k0 = blockIdx.y * 32;
  const int tx = threadIdx.x & 31;
  const int ty = threadIdx.x >> 5;
#pragma unroll
  for (int i = 0; i < 32; i += 8)
    tile[ty + i][tx] = (bf16_t)W[(size_t)(k0 + ty + i) * N + n0 + tx];
  __syncthreads();
#pragma unroll
  for (int i = 0; i < 32; i += 8)
    Wt[(size_t)(n0 + ty + i) * K + k0 + tx] = tile[tx][ty + i];
}

// ---------------------------------------------------------------------------
// GEMM: C[M][N] = A[M][K](bf16,row) * Bt[N][K](bf16, B transposed)
// 128x128 block tile, 8 waves of 32x64, K-step 64; tiles staged via TDM.
// ---------------------------------------------------------------------------
template <typename OutT>
__global__ __launch_bounds__(256) void gemm_bf16(
    const bf16_t* __restrict__ A, const bf16_t* __restrict__ Bt,
    OutT* __restrict__ C, int M, int N, int K) {
  __shared__ __align__(16) bf16_t Asm[128 * 64];
  __shared__ __align__(16) bf16_t Bsm[128 * 64];

  const int tid  = threadIdx.x;
  const int lane = tid & 31;
  const int wave = tid >> 5;
  const int wm = wave & 3;
  const int wn = wave >> 2;
  const int m0 = blockIdx.y * 128;
  const int n0 = blockIdx.x * 128;
  const int ln = lane & 15;
  const int hi = lane >> 4;

  const v8f vzero = {0.f, 0.f, 0.f, 0.f, 0.f, 0.f, 0.f, 0.f};
  v8f acc[2][4];
#pragma unroll
  for (int i = 0; i < 2; ++i)
#pragma unroll
    for (int j = 0; j < 4; ++j) acc[i][j] = vzero;

  const int sr = tid >> 1;                       // used by prefetch
  const int sc = (tid & 1) << 5;

  for (int kb = 0; kb < K; kb += 64) {
    // ---- TDM: DMA both 128x64 tiles straight into LDS (wave 0 issues) ----
    if (wave == 0) {
      tdm_load_2d(Asm, A + (size_t)m0 * K + kb, 64u, 128u, (unsigned)K);
      tdm_load_2d(Bsm, Bt + (size_t)n0 * K + kb, 64u, 128u, (unsigned)K);
      __builtin_amdgcn_s_wait_tensorcnt(0);
    }
    if (kb + 64 < K) {                            // warm L2 for next tile
      __builtin_prefetch(A  + (size_t)(m0 + sr) * K + kb + 64 + sc, 0, 0);
      __builtin_prefetch(Bt + (size_t)(n0 + sr) * K + kb + 64 + sc, 0, 0);
    }
    __syncthreads();
#pragma unroll
    for (int ks = 0; ks < 64; ks += 32) {
      v16bf af[2];
#pragma unroll
      for (int i = 0; i < 2; ++i)
        af[i] = load_frag_a(Asm + (wm * 32 + i * 16) * 64 + ks, 64, lane);
#pragma unroll
      for (int j = 0; j < 4; ++j) {
        v16bf bfr = load_frag_b(Bsm + (wn * 64 + j * 16) * 64 + ks, 64, lane);
#pragma unroll
        for (int i = 0; i < 2; ++i)
          acc[i][j] = wmma_bf16(af[i], bfr, acc[i][j]);
      }
    }
    __syncthreads();
  }
#pragma unroll
  for (int i = 0; i < 2; ++i)
#pragma unroll
    for (int j = 0; j < 4; ++j) {
      const int col = n0 + wn * 64 + j * 16 + ln;
#pragma unroll
      for (int jr = 0; jr < 8; ++jr) {
        const int row = m0 + wm * 32 + i * 16 + jr + 8 * hi;
        C[(size_t)row * N + col] = (OutT)acc[i][j][jr];
      }
    }
}

// ---------------------------------------------------------------------------
// Flash attention, causal. One block per (128 query rows, head, batch).
// K tile staged via TDM; V tile transposed in LDS by lanes (overlaps TDM).
// ---------------------------------------------------------------------------
__global__ __launch_bounds__(256) void flash_attn(
    const bf16_t* __restrict__ Q, const bf16_t* __restrict__ K,
    const bf16_t* __restrict__ V, bf16_t* __restrict__ O) {
  __shared__ __align__(16) bf16_t Ksm [64 * 128];   // [key][d]
  __shared__ __align__(16) bf16_t Vtsm[128 * 64];   // [d][key]
  __shared__ __align__(16) bf16_t Psm [8 * 16 * 64];

  const int tid  = threadIdx.x;
  const int lane = tid & 31;
  const int wave = tid >> 5;
  const int ln   = lane & 15;
  const int hi   = lane >> 4;
  const int m0   = blockIdx.x * 128;
  const int h    = blockIdx.y;
  const int b    = blockIdx.z;
  const int row0 = m0 + wave * 16;

  const size_t headoff = (size_t)h * HD_;
  const size_t rowbase = (size_t)b * S_;
  const float  SCALE   = 0.08838834764831845f;

  const bf16_t* qptr = Q + (rowbase + row0) * (size_t)H_ + headoff;
  v16bf qf[4];
#pragma unroll
  for (int kq = 0; kq < 4; ++kq)
    qf[kq] = load_frag_a(qptr + kq * 32, H_, lane);

  const v8f vzero = {0.f, 0.f, 0.f, 0.f, 0.f, 0.f, 0.f, 0.f};
  v8f acc[8];
  float m[8], l[8];
#pragma unroll
  for (int nb = 0; nb < 8; ++nb) acc[nb] = vzero;
#pragma unroll
  for (int jr = 0; jr < 8; ++jr) { m[jr] = -1e30f; l[jr] = 0.f; }

  const int nt = 2 * blockIdx.x + 2;
  for (int it = 0; it < nt; ++it) {
    const int t0 = it * 64;
    // ---- K tile via TDM (64 rows x 128 d, stride H) ----
    if (wave == 0)
      tdm_load_2d(Ksm, K + (rowbase + t0) * (size_t)H_ + headoff,
                  (unsigned)HD_, 64u, (unsigned)H_);
    // ---- V tile: per-lane load + LDS transpose (overlaps the DMA) ----
    for (int i = tid; i < 1024; i += 256) {
      const int r  = i >> 4;
      const int c8 = (i & 15) << 3;
      v8bf vv = *(const v8bf*)(V + (rowbase + t0 + r) * (size_t)H_ + headoff + c8);
#pragma unroll
      for (int e = 0; e < 8; ++e) Vtsm[(c8 + e) * 64 + r] = vv[e];
    }
    if (wave == 0) __builtin_amdgcn_s_wait_tensorcnt(0);
    __syncthreads();

    // ---- S = Q * K^T ----
    v8f s[4];
#pragma unroll
    for (int nb2 = 0; nb2 < 4; ++nb2) {
      s[nb2] = vzero;
#pragma unroll
      for (int kq = 0; kq < 4; ++kq) {
        v16bf kfr = load_frag_b(Ksm + (nb2 * 16) * 128 + kq * 32, 128, lane);
        s[nb2] = wmma_bf16(qf[kq], kfr, s[nb2]);
      }
    }

    // ---- scale + causal mask ----
#pragma unroll
    for (int nb2 = 0; nb2 < 4; ++nb2)
#pragma unroll
      for (int jr = 0; jr < 8; ++jr) {
        const int rg = row0 + jr + 8 * hi;
        const int cg = t0 + nb2 * 16 + ln;
        const float v = s[nb2][jr] * SCALE;
        s[nb2][jr] = (cg <= rg) ? v : -1e10f;
      }

    // ---- online softmax ----
#pragma unroll
    for (int jr = 0; jr < 8; ++jr) {
      float mx = fmaxf(fmaxf(s[0][jr], s[1][jr]), fmaxf(s[2][jr], s[3][jr]));
#pragma unroll
      for (int off = 1; off < 16; off <<= 1)
        mx = fmaxf(mx, __shfl_xor(mx, off, 32));
      const float mn   = fmaxf(m[jr], mx);
      const float corr = __expf(m[jr] - mn);
      m[jr] = mn;
      float ps = 0.f;
#pragma unroll
      for (int nb2 = 0; nb2 < 4; ++nb2) {
        const float pv = __expf(s[nb2][jr] - mn);
        s[nb2][jr] = pv;
        ps += pv;
      }
#pragma unroll
      for (int off = 1; off < 16; off <<= 1)
        ps += __shfl_xor(ps, off, 32);
      l[jr] = l[jr] * corr + ps;
#pragma unroll
      for (int nb = 0; nb < 8; ++nb) acc[nb][jr] *= corr;
    }

    // ---- P -> per-wave LDS (relayout C-frag -> A-frag) ----
    bf16_t* prow = Psm + wave * (16 * 64);
#pragma unroll
    for (int nb2 = 0; nb2 < 4; ++nb2)
#pragma unroll
      for (int jr = 0; jr < 8; ++jr)
        prow[(jr + 8 * hi) * 64 + nb2 * 16 + ln] = (bf16_t)s[nb2][jr];

    // ---- O += P(16x64) * V(64x128) ----
#pragma unroll
    for (int ks = 0; ks < 64; ks += 32) {
      v16bf pa = load_frag_a(prow + ks, 64, lane);
#pragma unroll
      for (int nb = 0; nb < 8; ++nb) {
        v16bf vb = load_frag_b(Vtsm + (nb * 16) * 64 + ks, 64, lane);
        acc[nb] = wmma_bf16(pa, vb, acc[nb]);
      }
    }
    __syncthreads();
  }

  // ---- normalize + store ----
#pragma unroll
  for (int jr = 0; jr < 8; ++jr) {
    const float inv = 1.0f / l[jr];
    const int rl = jr + 8 * hi;
    bf16_t* orow = O + (rowbase + row0 + rl) * (size_t)H_ + headoff;
#pragma unroll
    for (int nb = 0; nb < 8; ++nb)
      orow[nb * 16 + ln] = (bf16_t)(acc[nb][jr] * inv);
  }
}

// ---------------------------------------------------------------------------
extern "C" void kernel_launch(void* const* d_in, const int* in_sizes, int n_in,
                              void* d_out, int out_size, void* d_ws, size_t ws_size,
                              hipStream_t stream) {
  (void)in_sizes; (void)n_in; (void)out_size; (void)ws_size;
  const float* x  = (const float*)d_in[0];
  const float* Wq = (const float*)d_in[2];
  const float* Wk = (const float*)d_in[3];
  const float* Wv = (const float*)d_in[4];
  const float* Wo = (const float*)d_in[5];
  float* out = (float*)d_out;

  const size_t MS = (size_t)B_ * S_;
  bf16_t* p   = (bf16_t*)d_ws;
  bf16_t* xb  = p; p += MS * H_;
  bf16_t* Wqt = p; p += (size_t)H_ * H_;
  bf16_t* Wkt = p; p += (size_t)H_ * H_;
  bf16_t* Wvt = p; p += (size_t)H_ * H_;
  bf16_t* Wot = p; p += (size_t)H_ * H_;
  bf16_t* Qb  = p; p += MS * H_;
  bf16_t* Kb  = p; p += MS * H_;
  bf16_t* Vb  = p; p += MS * H_;
  bf16_t* Ob  = p; p += MS * H_;

  const int n4 = (int)(MS * H_ / 4);
  convert_bf16x4<<<n4 / 256, 256, 0, stream>>>((const float4*)x, xb, n4);

  dim3 tg(H_ / 32, H_ / 32);
  convert_transpose<<<tg, 256, 0, stream>>>(Wq, Wqt, H_, H_);
  convert_transpose<<<tg, 256, 0, stream>>>(Wk, Wkt, H_, H_);
  convert_transpose<<<tg, 256, 0, stream>>>(Wv, Wvt, H_, H_);
  convert_transpose<<<tg, 256, 0, stream>>>(Wo, Wot, H_, H_);

  dim3 gg(H_ / 128, (int)MS / 128);
  gemm_bf16<bf16_t><<<gg, 256, 0, stream>>>(xb, Wqt, Qb, (int)MS, H_, H_);
  gemm_bf16<bf16_t><<<gg, 256, 0, stream>>>(xb, Wkt, Kb, (int)MS, H_, H_);
  gemm_bf16<bf16_t><<<gg, 256, 0, stream>>>(xb, Wvt, Vb, (int)MS, H_, H_);

  dim3 fg(S_ / 128, NH_, B_);
  flash_attn<<<fg, 256, 0, stream>>>(Qb, Kb, Vb, Ob);

  gemm_bf16<float><<<gg, 256, 0, stream>>>(Ob, Wot, out, (int)MS, H_, H_);
}
